// SemanticLoss_17875653886443
// MI455X (gfx1250) — matrix operand order repaired and checked by reading, not compiled, not run on hardware
//
#include <hip/hip_runtime.h>
#include <hip/hip_bf16.h>
#include <stdint.h>

typedef __attribute__((ext_vector_type(16))) _Float16 v16h;
typedef __attribute__((ext_vector_type(8)))  _Float16 v8h;
typedef __attribute__((ext_vector_type(8)))  float    v8f;

#define N_ROWS   131072
#define DFEAT    256
#define N_CLASS  100
#define C_PAD    112                 // 7 tiles of 16
#define NUM_WG   128
#define CHUNK    (N_ROWS / NUM_WG)   // 1024
#define KSTEPS   (CHUNK / 32)        // 32
#define PADK     40                  // halves per FT row (80B, 16B-aligned, bank-friendly)
#define PADA     40                  // halves per A row  (80B, 16B-aligned, bank-friendly)
#define A_U4     ((C_PAD * PADA) / 8)   // 560 uint4 words in the A tile

static __device__ __forceinline__ uint16_t h2u(_Float16 h) {
  union { _Float16 h; uint16_t u; } x; x.h = h; return x.u;
}
static __device__ __forceinline__ _Float16 u2h(uint16_t u) {
  union { uint16_t u; _Float16 h; } x; x.u = u; return x.h;
}
static __device__ __forceinline__ void store_pair(_Float16* dst, float a, float b) {
  uint32_t u = (uint32_t)h2u((_Float16)a) | ((uint32_t)h2u((_Float16)b) << 16);
  *reinterpret_cast<uint32_t*>(dst) = u;
}

// ---------------- Stage 0: zero the count table ----------------
__global__ void k_init(uint32_t* __restrict__ counts) {
  counts[threadIdx.x] = 0u;   // 256 entries (2 matrices x 128 padded classes)
}

// ---------------- Stage 1: per-row argmax / weight / counts ----------------
__global__ void k_labels(const float* __restrict__ y_s,
                         const float* __restrict__ y_t,
                         uint32_t* __restrict__ lw,
                         uint32_t* __restrict__ counts) {
  const int mat = blockIdx.y;
  const int row = blockIdx.x * 256 + threadIdx.x;
  const float* yr = (mat ? y_t : y_s) + (size_t)row * N_CLASS;
  const float4* y4 = reinterpret_cast<const float4*>(yr);  // 400B rows, 16B aligned

  float best = -INFINITY; int bi = 0;
#pragma unroll
  for (int q = 0; q < 25; ++q) {
    float4 v = y4[q];
    if (v.x > best) { best = v.x; bi = 4 * q + 0; }
    if (v.y > best) { best = v.y; bi = 4 * q + 1; }
    if (v.z > best) { best = v.z; bi = 4 * q + 2; }
    if (v.w > best) { best = v.w; bi = 4 * q + 3; }
  }
  float w = (best > 0.9f) ? best : 0.0f;
  uint32_t packed = ((uint32_t)bi << 16) | (uint32_t)h2u((_Float16)w);
  lw[(size_t)mat * N_ROWS + row] = packed;
  atomicAdd(&counts[mat * 128 + bi], 1u);
}

// ---------------- Stage 2: WMMA weighted scatter-GEMM ----------------
// sums[c, f] = sum_i w_i * (label_i == c) * F[i, f]
// A (one-hot-weighted, [112 x 32] f16) built in LDS once per K-step;
// fragments via ds_load_b128, software-pipelined ahead of the WMMA burst.
__global__ __launch_bounds__(256)
void k_wmma(const float* __restrict__ s_feat,
            const float* __restrict__ t_feat,
            const uint32_t* __restrict__ lw,
            float* __restrict__ partials) {
  const int mat  = blockIdx.y;
  const int wg   = blockIdx.x;
  const float* F = mat ? t_feat : s_feat;
  const uint32_t* lwm = lw + (size_t)mat * N_ROWS;
  float* part = partials + ((size_t)(mat * NUM_WG + wg)) * (C_PAD * DFEAT);

  __shared__ __align__(16) _Float16 FT[DFEAT * PADK];   // transposed f16 tile [n][k]
  __shared__ __align__(16) _Float16 AH[C_PAD * PADA];   // one-hot weighted A [class][k]

  const int tid  = threadIdx.x;
  const int lane = tid & 31;
  const int wave = tid >> 5;
  const int g    = lane >> 4;        // half-wave group
  const int m    = lane & 15;        // M (class-in-tile) for A, N for B/C/D
  const int kb   = 16 * g;           // B fragment K base (contiguous K order, same as A)
  const int n0   = wave * 32 + m;    // this wave's feature columns: tile0
  const int n1   = n0 + 16;          // tile1

  v8f acc[7][2];
#pragma unroll
  for (int i = 0; i < 7; ++i) { acc[i][0] = {}; acc[i][1] = {}; }

  for (int ks = 0; ks < KSTEPS; ++ks) {
    const int rbase = wg * CHUNK + ks * 32;
    __syncthreads();   // previous iteration's LDS reads complete

    // ---- zero A tile: branch-free (560 uint4 = 2 full strides + 48 tail) ----
    {
      uint4* az = reinterpret_cast<uint4*>(AH);
      const uint4 z = {0u, 0u, 0u, 0u};
      az[tid]       = z;
      az[tid + 256] = z;
      if (tid < A_U4 - 512) az[tid + 512] = z;
    }

    // ---- stage 32 rows x 256 cols f32 -> LDS f16 transposed (packed b32 stores) ----
    const float4* src = reinterpret_cast<const float4*>(F + (size_t)rbase * DFEAT);
#pragma unroll
    for (int q = 0; q < 4; ++q) {
      const int p  = tid + q * 256;        // 0..1023: (k-pair, col4)
      const int k2 = p >> 6;               // 0..15 -> rows 2k2, 2k2+1
      const int c4 = (p & 63) << 2;
      float4 v0 = src[(2 * k2 + 0) * 64 + (p & 63)];
      float4 v1 = src[(2 * k2 + 1) * 64 + (p & 63)];
      store_pair(&FT[(c4 + 0) * PADK + 2 * k2], v0.x, v1.x);
      store_pair(&FT[(c4 + 1) * PADK + 2 * k2], v0.y, v1.y);
      store_pair(&FT[(c4 + 2) * PADK + 2 * k2], v0.z, v1.z);
      store_pair(&FT[(c4 + 3) * PADK + 2 * k2], v0.w, v1.w);
    }
    if (ks + 1 < KSTEPS)  // hint next 32-row tile toward the caches
      __builtin_prefetch(&src[2048 + tid * 8], 0, 1);
    __syncthreads();

    // ---- scatter one-hot weights into A ----
    if (tid < 32) {
      const uint32_t p = lwm[rbase + tid];
      AH[(p >> 16) * PADA + tid] = u2h((uint16_t)p);
    }
    __syncthreads();

    // ---- B fragments: 16 contiguous f16 per lane from transposed LDS ----
    v8h b0lo = *reinterpret_cast<const v8h*>(&FT[n0 * PADK + kb]);
    v8h b0hi = *reinterpret_cast<const v8h*>(&FT[n0 * PADK + kb + 8]);
    v8h b1lo = *reinterpret_cast<const v8h*>(&FT[n1 * PADK + kb]);
    v8h b1hi = *reinterpret_cast<const v8h*>(&FT[n1 * PADK + kb + 8]);
    v16h b0 = __builtin_shufflevector(b0lo, b0hi, 0,1,2,3,4,5,6,7,8,9,10,11,12,13,14,15);
    v16h b1 = __builtin_shufflevector(b1lo, b1hi, 0,1,2,3,4,5,6,7,8,9,10,11,12,13,14,15);

    // ---- software-pipeline: issue all 7 A-fragment loads first ----
    v16h afrag[7];
#pragma unroll
    for (int mt = 0; mt < 7; ++mt) {
      const int cls = mt * 16 + m;
      // A fragment: K 16g..16g+15 contiguous, matching B fragment order
      v8h alo = *reinterpret_cast<const v8h*>(&AH[cls * PADA + 16 * g]);
      v8h ahi = *reinterpret_cast<const v8h*>(&AH[cls * PADA + 16 * g + 8]);
      afrag[mt] = __builtin_shufflevector(alo, ahi,
                                          0,1,2,3,4,5,6,7,8,9,10,11,12,13,14,15);
    }

    // ---- 14 back-to-back WMMAs ----
#pragma unroll
    for (int mt = 0; mt < 7; ++mt) {
      acc[mt][0] = __builtin_amdgcn_wmma_f32_16x16x32_f16(
          false, afrag[mt], false, b0, (short)0, acc[mt][0], false, false);
      acc[mt][1] = __builtin_amdgcn_wmma_f32_16x16x32_f16(
          false, afrag[mt], false, b1, (short)0, acc[mt][1], false, false);
    }
  }

  // ---- write per-WG partial sums: disjoint (class, feature) per lane ----
#pragma unroll
  for (int mt = 0; mt < 7; ++mt)
#pragma unroll
    for (int nt = 0; nt < 2; ++nt)
#pragma unroll
      for (int r = 0; r < 8; ++r) {
        const int cls  = mt * 16 + r + 8 * g;          // C/D layout: VGPR r -> M=r (+8 hi half)
        const int feat = wave * 32 + nt * 16 + m;
        part[cls * DFEAT + feat] = acc[mt][nt][r];
      }
}

// ---------------- Stage 3: reduce partials, EMA, per-class sq-diff ----------------
__global__ void k_final(const float* __restrict__ partials,
                        const uint32_t* __restrict__ counts,
                        const float* __restrict__ s_cent,
                        const float* __restrict__ t_cent,
                        float* __restrict__ loss_part) {
  const int c = blockIdx.x;      // 0..99
  const int j = threadIdx.x;     // 0..255
  float s_sum = 0.0f, t_sum = 0.0f;
  for (int wg = 0; wg < NUM_WG; ++wg) {
    s_sum += partials[((size_t)(wg)          * C_PAD + c) * DFEAT + j];
    t_sum += partials[((size_t)(NUM_WG + wg) * C_PAD + c) * DFEAT + j];
  }
  const float cs_n = (float)max(counts[c], 1u);
  const float ct_n = (float)max(counts[128 + c], 1u);
  const float cs = 0.7f * s_cent[c * DFEAT + j] + 0.3f * (s_sum / cs_n);
  const float ct = 0.7f * t_cent[c * DFEAT + j] + 0.3f * (t_sum / ct_n);
  const float d  = cs - ct;

  __shared__ float red[256];
  red[j] = d * d;
  __syncthreads();
  for (int off = 128; off > 0; off >>= 1) {
    if (j < off) red[j] += red[j + off];
    __syncthreads();
  }
  if (j == 0) loss_part[c] = red[0];
}

// ---------------- Stage 4: deterministic final fold ----------------
__global__ void k_reduce(const float* __restrict__ loss_part, float* __restrict__ out) {
  __shared__ float s[128];
  const int t = threadIdx.x;
  s[t] = (t < N_CLASS) ? loss_part[t] : 0.0f;
  __syncthreads();
  for (int off = 64; off > 0; off >>= 1) {
    if (t < off) s[t] += s[t + off];
    __syncthreads();
  }
  if (t == 0) out[0] = s[0] * (1.0f / (float)(N_CLASS * DFEAT));
}

extern "C" void kernel_launch(void* const* d_in, const int* in_sizes, int n_in,
                              void* d_out, int out_size, void* d_ws, size_t ws_size,
                              hipStream_t stream) {
  const float* s_feat = (const float*)d_in[0];
  const float* t_feat = (const float*)d_in[1];
  const float* y_s    = (const float*)d_in[2];
  const float* y_t    = (const float*)d_in[3];
  const float* s_cent = (const float*)d_in[4];
  const float* t_cent = (const float*)d_in[5];
  float* out = (float*)d_out;

  char* ws = (char*)d_ws;
  constexpr size_t OFF_LW   = 0;                                   // 2*N u32
  constexpr size_t OFF_CNT  = OFF_LW + (size_t)2 * N_ROWS * 4;     // 256 u32
  constexpr size_t OFF_LOSS = OFF_CNT + 256 * 4;                   // 128 f32
  constexpr size_t OFF_PART = (OFF_LOSS + 128 * 4 + 255) & ~(size_t)255;

  uint32_t* lw      = (uint32_t*)(ws + OFF_LW);
  uint32_t* counts  = (uint32_t*)(ws + OFF_CNT);
  float*    lossbuf = (float*)(ws + OFF_LOSS);
  float*    parts   = (float*)(ws + OFF_PART);   // 2*NUM_WG*112*256 f32

  k_init<<<1, 256, 0, stream>>>(counts);
  k_labels<<<dim3(N_ROWS / 256, 2), 256, 0, stream>>>(y_s, y_t, lw, counts);
  k_wmma<<<dim3(NUM_WG, 2), 256, 0, stream>>>(s_feat, t_feat, lw, parts);
  k_final<<<N_CLASS, 256, 0, stream>>>(parts, counts, s_cent, t_cent, lossbuf);
  k_reduce<<<1, 128, 0, stream>>>(lossbuf, out);
}
